// SwAV_71365176590738
// MI455X (gfx1250) — matrix operand order repaired and checked by reading, not compiled, not run on hardware
//
#include <hip/hip_runtime.h>
#include <hip/hip_bf16.h>

// ---------------------------------------------------------------------------
// Problem constants
// ---------------------------------------------------------------------------
#define BATCH    256
#define DIM      256
#define NPROTO   3072
#define NQUEUE   32768
#define NTOT     (BATCH + NQUEUE)      // 33024
#define LAMBDA_  20.0f
#define NITERS   10
#define NSPLIT   129                   // 129 * 256 == 33024 row chunks for col pass

typedef __attribute__((ext_vector_type(8)))  __bf16 v8bf;
typedef __attribute__((ext_vector_type(16))) __bf16 v16bf;
typedef __attribute__((ext_vector_type(8)))  float  v8f;

// ---------------------------------------------------------------------------
// bf16 bit helpers (round-to-nearest-even), no __bf16 arithmetic needed
// ---------------------------------------------------------------------------
__device__ __forceinline__ __bf16 f2bf(float x) {
    unsigned u = __float_as_uint(x);
    unsigned r = u + 0x7FFFu + ((u >> 16) & 1u);
    unsigned short s = (unsigned short)(r >> 16);
    return __builtin_bit_cast(__bf16, s);
}
__device__ __forceinline__ float bf2f(__bf16 h) {
    unsigned short s = __builtin_bit_cast(unsigned short, h);
    return __uint_as_float(((unsigned)s) << 16);
}

// ---------------------------------------------------------------------------
// 1) Row L2-normalize + split into bf16 hi/lo. One wave32 per row, D=256.
// ---------------------------------------------------------------------------
__global__ void normsplit_kernel(const float* __restrict__ X,
                                 __bf16* __restrict__ Hi,
                                 __bf16* __restrict__ Lo) {
    const int wave = threadIdx.x >> 5, lane = threadIdx.x & 31;
    const long row = (long)blockIdx.x * 8 + wave;
    const float* x = X + row * DIM;
    float vals[8];
    float ss = 0.f;
#pragma unroll
    for (int j = 0; j < 8; ++j) {
        float v = x[lane + 32 * j];
        vals[j] = v;
        ss += v * v;
    }
#pragma unroll
    for (int off = 16; off; off >>= 1) ss += __shfl_xor(ss, off, 32);
    const float rn = 1.0f / sqrtf(ss);
#pragma unroll
    for (int j = 0; j < 8; ++j) {
        float v  = vals[j] * rn;
        __bf16 h = f2bf(v);
        __bf16 l = f2bf(v - bf2f(h));
        Hi[row * DIM + lane + 32 * j] = h;
        Lo[row * DIM + lane + 32 * j] = l;
    }
}

// ---------------------------------------------------------------------------
// 2) WMMA GEMM (split-bf16 x3) with fused exp(20*p) epilogue.
//    C[m][n] = dot(A_row m, B_row n)  (B rows are Qn rows since B = Qn^T).
//    One 16x16 tile per wave, 4 waves per block.
// ---------------------------------------------------------------------------
__device__ __forceinline__ v16bf load_a16(const __bf16* rowp, int k0, int half) {
    // A 16x32 bf16: per-lane elems K = {b..b+7} U {b+16..b+23}, b = k0 + half*8
    const int b = k0 + half * 8;
    v8bf c0 = *(const v8bf*)(rowp + b);
    v8bf c1 = *(const v8bf*)(rowp + b + 16);
    return __builtin_shufflevector(c0, c1, 0,1,2,3,4,5,6,7,8,9,10,11,12,13,14,15);
}
__device__ __forceinline__ v16bf load_b16(const __bf16* rowp, int k0, int half) {
    // B 32x16 bf16: per-lane column = lane&15, K = {b..b+7} U {b+16..b+23}, b = k0 + half*8
    const int b = k0 + half * 8;
    v8bf c0 = *(const v8bf*)(rowp + b);
    v8bf c1 = *(const v8bf*)(rowp + b + 16);
    return __builtin_shufflevector(c0, c1, 0,1,2,3,4,5,6,7,8,9,10,11,12,13,14,15);
}

__global__ void gemm_exp_kernel(const __bf16* __restrict__ Ahi,
                                const __bf16* __restrict__ Alo,
                                const __bf16* __restrict__ Bhi,
                                const __bf16* __restrict__ Blo,
                                float* __restrict__ Pout,   // may be null
                                float* __restrict__ Kout) {
    const int lane = threadIdx.x & 31;
    const int wave = threadIdx.x >> 5;
    const int half = lane >> 4;
    const int r16  = lane & 15;
    const int m0 = blockIdx.y * 16;
    const int n0 = (blockIdx.x * 4 + wave) * 16;

    const __bf16* ah = Ahi + (long)(m0 + r16) * DIM;
    const __bf16* al = Alo + (long)(m0 + r16) * DIM;
    const __bf16* bh = Bhi + (long)(n0 + r16) * DIM;
    const __bf16* bl = Blo + (long)(n0 + r16) * DIM;

    v8f acc = {};
#pragma unroll
    for (int kk = 0; kk < 8; ++kk) {
        const int k0 = kk * 32;
        v16bf a_h = load_a16(ah, k0, half);
        v16bf a_l = load_a16(al, k0, half);
        v16bf b_h = load_b16(bh, k0, half);
        v16bf b_l = load_b16(bl, k0, half);
        acc = __builtin_amdgcn_wmma_f32_16x16x32_bf16(false, a_h, false, b_h,
                                                      (short)0, acc, false, false);
        acc = __builtin_amdgcn_wmma_f32_16x16x32_bf16(false, a_l, false, b_h,
                                                      (short)0, acc, false, false);
        acc = __builtin_amdgcn_wmma_f32_16x16x32_bf16(false, a_h, false, b_l,
                                                      (short)0, acc, false, false);
    }

    const int gn = n0 + r16;
#pragma unroll
    for (int r = 0; r < 8; ++r) {
        const long gm = m0 + r + half * 8;
        const float val = acc[r];
        if (Pout) Pout[gm * NPROTO + gn] = val;
        Kout[gm * NPROTO + gn] = __expf(LAMBDA_ * val);
    }
}

// ---------------------------------------------------------------------------
// 3) Sinkhorn state init: u = 1 (33024 x2 views), v = 1 (3072 x2 views)
// ---------------------------------------------------------------------------
__global__ void init_uv_kernel(float* u1, float* u2, float* v1, float* v2) {
    const int i = blockIdx.x * 256 + threadIdx.x;
    if (i < NTOT)   { u1[i] = 1.f; u2[i] = 1.f; }
    if (i < NPROTO) { v1[i] = 1.f; v2[i] = 1.f; }
}

// ---------------------------------------------------------------------------
// 4) Row pass: u = 1 / (K v) for BOTH views with a single read of shared Kq.
//    One wave per row; float4 streaming loads.
// ---------------------------------------------------------------------------
__global__ void sinkhorn_rows_kernel(const float* __restrict__ Kq,
                                     const float* __restrict__ Kz1,
                                     const float* __restrict__ Kz2,
                                     const float* __restrict__ v1,
                                     const float* __restrict__ v2,
                                     float* __restrict__ u1,
                                     float* __restrict__ u2) {
    const int wave = threadIdx.x >> 5, lane = threadIdx.x & 31;
    const int t = blockIdx.x * 8 + wave;
    const float4* vv1 = (const float4*)v1;
    const float4* vv2 = (const float4*)v2;
    float acc1 = 0.f, acc2 = 0.f;
    if (t < BATCH) {
        const float4* r1 = (const float4*)(Kz1 + (long)t * NPROTO);
        const float4* r2 = (const float4*)(Kz2 + (long)t * NPROTO);
        for (int j = lane; j < NPROTO / 4; j += 32) {
            float4 a = r1[j], b = vv1[j];
            acc1 += a.x * b.x + a.y * b.y + a.z * b.z + a.w * b.w;
            float4 c = r2[j], d = vv2[j];
            acc2 += c.x * d.x + c.y * d.y + c.z * d.z + c.w * d.w;
        }
    } else {
        const float4* r = (const float4*)(Kq + (long)(t - BATCH) * NPROTO);
        for (int j = lane; j < NPROTO / 4; j += 32) {
            float4 a = r[j], b = vv1[j], d = vv2[j];
            acc1 += a.x * b.x + a.y * b.y + a.z * b.z + a.w * b.w;
            acc2 += a.x * d.x + a.y * d.y + a.z * d.z + a.w * d.w;
        }
    }
#pragma unroll
    for (int off = 16; off; off >>= 1) {
        acc1 += __shfl_xor(acc1, off, 32);
        acc2 += __shfl_xor(acc2, off, 32);
    }
    if (lane == 0) { u1[t] = 1.f / acc1; u2[t] = 1.f / acc2; }
}

// ---------------------------------------------------------------------------
// 5) Column pass, split-K partials: part[s][col] = sum_{t in chunk s} K[t][col]*u[t]
//    grid = (NPROTO/256, NSPLIT), 256-row chunks; chunk 0 is exactly the z rows.
// ---------------------------------------------------------------------------
__global__ void sinkhorn_cols_partial_kernel(const float* __restrict__ Kq,
                                             const float* __restrict__ Kz1,
                                             const float* __restrict__ Kz2,
                                             const float* __restrict__ u1,
                                             const float* __restrict__ u2,
                                             float* __restrict__ p1,
                                             float* __restrict__ p2) {
    const int col = blockIdx.x * 256 + threadIdx.x;
    const int t0  = blockIdx.y * 256;
    float a1 = 0.f, a2 = 0.f;
    if (t0 == 0) {
        const float* b1 = Kz1 + col;
        const float* b2 = Kz2 + col;
#pragma unroll 4
        for (int i = 0; i < 256; ++i) {
            a1 += b1[(long)i * NPROTO] * u1[i];
            a2 += b2[(long)i * NPROTO] * u2[i];
        }
    } else {
        const float* b = Kq + (long)(t0 - BATCH) * NPROTO + col;
        const float* uu1 = u1 + t0;
        const float* uu2 = u2 + t0;
#pragma unroll 4
        for (int i = 0; i < 256; ++i) {
            const float k = b[(long)i * NPROTO];
            a1 += k * uu1[i];
            a2 += k * uu2[i];
        }
    }
    p1[(long)blockIdx.y * NPROTO + col] = a1;
    p2[(long)blockIdx.y * NPROTO + col] = a2;
}

// 6) Column reduce: v[col] = c / sum_s part[s][col]
__global__ void sinkhorn_cols_reduce_kernel(const float* __restrict__ p1,
                                            const float* __restrict__ p2,
                                            float* __restrict__ v1,
                                            float* __restrict__ v2) {
    const int col = blockIdx.x * 256 + threadIdx.x;
    float s1 = 0.f, s2 = 0.f;
    for (int s = 0; s < NSPLIT; ++s) {
        s1 += p1[(long)s * NPROTO + col];
        s2 += p2[(long)s * NPROTO + col];
    }
    const float c = (float)NTOT / (float)NPROTO;   // 10.75
    v1[col] = c / s1;
    v2[col] = c / s2;
}

// ---------------------------------------------------------------------------
// Block reductions (256 threads)
// ---------------------------------------------------------------------------
__device__ __forceinline__ float blockSum256(float x, float* red) {
    const int tid = threadIdx.x;
    red[tid] = x; __syncthreads();
    for (int s = 128; s > 0; s >>= 1) {
        if (tid < s) red[tid] += red[tid + s];
        __syncthreads();
    }
    const float r = red[0]; __syncthreads();
    return r;
}
__device__ __forceinline__ float blockMax256(float x, float* red) {
    const int tid = threadIdx.x;
    red[tid] = x; __syncthreads();
    for (int s = 128; s > 0; s >>= 1) {
        if (tid < s) red[tid] = fmaxf(red[tid], red[tid + s]);
        __syncthreads();
    }
    const float r = red[0]; __syncthreads();
    return r;
}

// ---------------------------------------------------------------------------
// 7) Per-row loss: rowloss[m] = -(u2[m]*sum_n Kz2[m,n]*v2[n]*(P1[m,n]-lse1)
//                               + u1[m]*sum_n Kz1[m,n]*v1[n]*(P2[m,n]-lse2))
// ---------------------------------------------------------------------------
__global__ void loss_rows_kernel(const float* __restrict__ P1,
                                 const float* __restrict__ P2,
                                 const float* __restrict__ Kz1,
                                 const float* __restrict__ Kz2,
                                 const float* __restrict__ u1,
                                 const float* __restrict__ u2,
                                 const float* __restrict__ v1,
                                 const float* __restrict__ v2,
                                 float* __restrict__ rowloss) {
    __shared__ float red[256];
    const int m = blockIdx.x, tid = threadIdx.x;
    const float* p1 = P1  + (long)m * NPROTO;
    const float* p2 = P2  + (long)m * NPROTO;
    const float* k1 = Kz1 + (long)m * NPROTO;
    const float* k2 = Kz2 + (long)m * NPROTO;

    float mx1 = -3.402823466e38f, mx2 = -3.402823466e38f;
    for (int n = tid; n < NPROTO; n += 256) {
        mx1 = fmaxf(mx1, p1[n]);
        mx2 = fmaxf(mx2, p2[n]);
    }
    mx1 = blockMax256(mx1, red);
    mx2 = blockMax256(mx2, red);

    float s1 = 0.f, s2 = 0.f;
    for (int n = tid; n < NPROTO; n += 256) {
        s1 += expf(p1[n] - mx1);
        s2 += expf(p2[n] - mx2);
    }
    s1 = blockSum256(s1, red);
    s2 = blockSum256(s2, red);
    const float lse1 = mx1 + logf(s1);
    const float lse2 = mx2 + logf(s2);

    float d1 = 0.f, d2 = 0.f;
    for (int n = tid; n < NPROTO; n += 256) {
        d1 += k2[n] * v2[n] * (p1[n] - lse1);
        d2 += k1[n] * v1[n] * (p2[n] - lse2);
    }
    d1 = blockSum256(d1, red);
    d2 = blockSum256(d2, red);
    if (tid == 0) rowloss[m] = -(u2[m] * d1 + u1[m] * d2);
}

// 8) Final scalar: out[0] = mean(rowloss)  (both views already folded per row)
__global__ void final_reduce_kernel(const float* __restrict__ rowloss,
                                    float* __restrict__ out) {
    __shared__ float red[256];
    float s = blockSum256(rowloss[threadIdx.x], red);
    if (threadIdx.x == 0) out[0] = s * (1.0f / (float)BATCH);
}

// ---------------------------------------------------------------------------
// Host launcher
// ---------------------------------------------------------------------------
extern "C" void kernel_launch(void* const* d_in, const int* in_sizes, int n_in,
                              void* d_out, int out_size, void* d_ws, size_t ws_size,
                              hipStream_t stream) {
    const float* X1 = (const float*)d_in[0];   // [256,256]
    const float* X2 = (const float*)d_in[1];   // [256,256]
    const float* Q  = (const float*)d_in[2];   // [3072,256]
    const float* FQ = (const float*)d_in[3];   // [32768,256]
    float* out = (float*)d_out;

    // ---- carve workspace (all offsets 256B aligned) ----
    char* ws = (char*)d_ws;
    size_t off = 0;
    auto carve = [&](size_t bytes) -> void* {
        void* p = ws + off;
        off += (bytes + 255) & ~(size_t)255;
        return p;
    };
    __bf16* Qn_hi = (__bf16*)carve((size_t)NPROTO * DIM * 2);
    __bf16* Qn_lo = (__bf16*)carve((size_t)NPROTO * DIM * 2);
    __bf16* Z1_hi = (__bf16*)carve((size_t)BATCH  * DIM * 2);
    __bf16* Z1_lo = (__bf16*)carve((size_t)BATCH  * DIM * 2);
    __bf16* Z2_hi = (__bf16*)carve((size_t)BATCH  * DIM * 2);
    __bf16* Z2_lo = (__bf16*)carve((size_t)BATCH  * DIM * 2);
    __bf16* Zq_hi = (__bf16*)carve((size_t)NQUEUE * DIM * 2);
    __bf16* Zq_lo = (__bf16*)carve((size_t)NQUEUE * DIM * 2);
    float* P1   = (float*)carve((size_t)BATCH  * NPROTO * 4);
    float* P2   = (float*)carve((size_t)BATCH  * NPROTO * 4);
    float* Kz1  = (float*)carve((size_t)BATCH  * NPROTO * 4);
    float* Kz2  = (float*)carve((size_t)BATCH  * NPROTO * 4);
    float* Kq   = (float*)carve((size_t)NQUEUE * NPROTO * 4);   // 402 MB
    float* u1   = (float*)carve((size_t)NTOT * 4);
    float* u2   = (float*)carve((size_t)NTOT * 4);
    float* v1   = (float*)carve((size_t)NPROTO * 4);
    float* v2   = (float*)carve((size_t)NPROTO * 4);
    float* par1 = (float*)carve((size_t)NSPLIT * NPROTO * 4);
    float* par2 = (float*)carve((size_t)NSPLIT * NPROTO * 4);
    float* rowloss = (float*)carve((size_t)BATCH * 4);
    (void)ws_size; (void)in_sizes; (void)n_in; (void)out_size;

    // ---- 1) normalize + bf16 split ----
    normsplit_kernel<<<BATCH  / 8, 256, 0, stream>>>(X1, Z1_hi, Z1_lo);
    normsplit_kernel<<<BATCH  / 8, 256, 0, stream>>>(X2, Z2_hi, Z2_lo);
    normsplit_kernel<<<NPROTO / 8, 256, 0, stream>>>(Q,  Qn_hi, Qn_lo);
    normsplit_kernel<<<NQUEUE / 8, 256, 0, stream>>>(FQ, Zq_hi, Zq_lo);

    // ---- 2) WMMA GEMMs with exp epilogue ----
    {
        dim3 gq(NPROTO / 64, NQUEUE / 16);   // (48, 2048)
        gemm_exp_kernel<<<gq, 128, 0, stream>>>(Zq_hi, Zq_lo, Qn_hi, Qn_lo,
                                                nullptr, Kq);
        dim3 gz(NPROTO / 64, BATCH / 16);    // (48, 16)
        gemm_exp_kernel<<<gz, 128, 0, stream>>>(Z1_hi, Z1_lo, Qn_hi, Qn_lo, P1, Kz1);
        gemm_exp_kernel<<<gz, 128, 0, stream>>>(Z2_hi, Z2_lo, Qn_hi, Qn_lo, P2, Kz2);
    }

    // ---- 3) Sinkhorn (both views fused per pass over shared Kq) ----
    init_uv_kernel<<<NSPLIT, 256, 0, stream>>>(u1, u2, v1, v2);
    for (int it = 0; it < NITERS; ++it) {
        sinkhorn_rows_kernel<<<NTOT / 8, 256, 0, stream>>>(Kq, Kz1, Kz2,
                                                           v1, v2, u1, u2);
        dim3 gp(NPROTO / 256, NSPLIT);       // (12, 129)
        sinkhorn_cols_partial_kernel<<<gp, 256, 0, stream>>>(Kq, Kz1, Kz2,
                                                             u1, u2, par1, par2);
        sinkhorn_cols_reduce_kernel<<<NPROTO / 256, 256, 0, stream>>>(par1, par2,
                                                                      v1, v2);
    }

    // ---- 4) loss ----
    loss_rows_kernel<<<BATCH, 256, 0, stream>>>(P1, P2, Kz1, Kz2,
                                                u1, u2, v1, v2, rowloss);
    final_reduce_kernel<<<1, 256, 0, stream>>>(rowloss, out);
}
